// U_Mamba_block_69776038691242
// MI455X (gfx1250) — compile-verified, matrix operand order checked by reading, not run on hardware
//
#include <hip/hip_runtime.h>
#include <hip/hip_bf16.h>

// ---------------------------------------------------------------------------
// Mamba block for MI455X (gfx1250, wave32).
// All four big GEMMs (~172 GFLOP) run on v_wmma_f32_16x16x32_bf16 with
// pre-converted bf16 operands, LDS double-buffering, and ASYNC global->LDS
// staging (global_load_async_to_lds_b128, ASYNCcnt) overlapping the WMMAs.
// Block tile 128x128, 8 waves, wave tile 32x64 (2x4 WMMA tiles, 64 acc VGPRs).
// Conv / B,C projection / selective scan / gate+LN are VALU kernels.
// ---------------------------------------------------------------------------

#define USE_ASYNC_STAGE 1   // 1: async copy to LDS via inline asm; 0: reg pipeline

typedef __attribute__((ext_vector_type(16))) __bf16 v16bf;
typedef __attribute__((ext_vector_type(8)))  float  v8f;

#define TM   128        // workgroup tile M
#define TN   128        // workgroup tile N
#define TKK  32         // K depth staged in LDS per iteration (one WMMA K)
#define LDSS 40         // padded bf16 row stride (80 B, multiple of 16 B)

__device__ __forceinline__ float silu_f(float v) {
  return v * __frcp_rn(1.0f + __expf(-v));
}
__device__ __forceinline__ float softplus_clip_f(float v) {
  float sp = (v > 20.0f) ? v : __logf(1.0f + __expf(v));
  return fminf(fmaxf(sp, 1e-4f), 10.0f);
}

__device__ __forceinline__ uint4 ld16g(const __bf16* p) {
  return *reinterpret_cast<const uint4*>(p);
}
__device__ __forceinline__ void st16s(__bf16* p, uint4 v) {
  *reinterpret_cast<uint4*>(p) = v;
}

#if USE_ASYNC_STAGE
// Per-lane async copy of 16 B global -> LDS, tracked by ASYNCcnt.
// Generic LDS pointers carry the LDS byte offset in their low 32 bits
// (ISA flat-addressing: LDS aperture truncates to addr[31:0]).
__device__ __forceinline__ void async_g2l_b128(const __bf16* lds_dst,
                                               const __bf16* gsrc) {
  const unsigned  ldsoff = (unsigned)(uintptr_t)lds_dst;
  const unsigned long long ga = (unsigned long long)(uintptr_t)gsrc;
  asm volatile("global_load_async_to_lds_b128 %0, %1, off"
               :: "v"(ldsoff), "v"(ga) : "memory");
}
__device__ __forceinline__ void wait_asynccnt0() {
  asm volatile("s_wait_asynccnt 0x0" ::: "memory");
}
#endif

// Load one 16x32 bf16 fragment from LDS in the ISA per-lane layout:
// lanes 0-15  : row = lane,    K = {0..7, 16..23}
// lanes 16-31 : row = lane-16, K = {8..15, 24..31}
__device__ __forceinline__ v16bf load_frag(const __bf16* base, int lane) {
  const int half = (lane >> 4) & 1;
  const __bf16* p = base + (lane & 15) * LDSS + half * 8;
  union { uint4 u[2]; v16bf v; } f;
  f.u[0] = *reinterpret_cast<const uint4*>(p);
  f.u[1] = *reinterpret_cast<const uint4*>(p + 16);
  return f.v;
}

// Store one 16x16 f32 accumulator tile (lanes 0-15: N=lane, M=0..7 in v[0..7];
// lanes 16-31: N=lane-16, M=8..15), with optional epilogue + bf16 output.
template <int EPI, int OUTBF>  // EPI: 0=none, 1=softplus+clip, 2=silu
__device__ __forceinline__ void store_tile(void* __restrict__ Cout,
                                           const float* __restrict__ bias,
                                           v8f acc, int row0, int col0, int N,
                                           int lane) {
  const int col   = col0 + (lane & 15);
  const int rbase = row0 + ((lane >> 4) << 3);
  const float b = bias ? bias[col] : 0.0f;
#pragma unroll
  for (int i = 0; i < 8; ++i) {
    float v = acc[i] + b;
    if (EPI == 1) v = softplus_clip_f(v);
    if (EPI == 2) v = silu_f(v);
    if (OUTBF) ((__bf16*)Cout)[(size_t)(rbase + i) * N + col] = (__bf16)v;
    else       ((float*)Cout)[(size_t)(rbase + i) * N + col]  = v;
  }
}

#define WMMA_BF16(d, a, b) \
  d = __builtin_amdgcn_wmma_f32_16x16x32_bf16(false, a, false, b, (short)0, d, false, false)

// C[M,N] = epi(A[M,K] @ Bw[N,K]^T + bias).  M%128==0, N%128==0, K%32==0.
template <int EPI, int OUTBF>
__global__ __launch_bounds__(256, 1) void gemm_bf16_wmma(
    const __bf16* __restrict__ A, const __bf16* __restrict__ Bw,
    const float* __restrict__ bias, void* __restrict__ Cout,
    int M, int N, int K) {
  __shared__ __attribute__((aligned(16))) __bf16 As[2][TM * LDSS];
  __shared__ __attribute__((aligned(16))) __bf16 Bs[2][TN * LDSS];

  const int tid  = threadIdx.x;
  const int lane = tid & 31;
  const int wave = tid >> 5;          // 0..7
  const int wr   = (wave >> 1) * 32;  // wave row base  (0/32/64/96)
  const int wc   = (wave & 1) * 64;   // wave col base  (0/64)
  const int m0   = blockIdx.x * TM;
  const int n0   = blockIdx.y * TN;

  // Per-thread staging geometry (loop-invariant): tile is 128 rows x 32 K of
  // bf16 = 512 16B-chunks; each thread moves chunks {tid, tid+256} for A and B.
  const int ch0 = tid,        r0 = ch0 >> 2, k0c = (ch0 & 3) << 3;
  const int ch1 = tid + 256,  r1 = ch1 >> 2, k1c = (ch1 & 3) << 3;
  const __bf16* Ab = A  + (size_t)m0 * K;
  const __bf16* Bb = Bw + (size_t)n0 * K;
  const size_t gA0 = (size_t)r0 * K + k0c, gA1 = (size_t)r1 * K + k1c;
  const int    s0  = r0 * LDSS + k0c,      s1  = r1 * LDSS + k1c;

  v8f c00 = {}, c01 = {}, c02 = {}, c03 = {};
  v8f c10 = {}, c11 = {}, c12 = {}, c13 = {};

#if USE_ASYNC_STAGE
  // ---- async-staged pipeline: issue tile 0 copies directly into LDS[0]
  async_g2l_b128(&As[0][s0], Ab + gA0);
  async_g2l_b128(&As[0][s1], Ab + gA1);
  async_g2l_b128(&Bs[0][s0], Bb + gA0);
  async_g2l_b128(&Bs[0][s1], Bb + gA1);

  int p = 0;
  for (int k0 = 0; k0 < K; k0 += TKK) {
    wait_asynccnt0();     // this wave's copies for tile k are in LDS
    __syncthreads();      // everyone's copies done; compute(k-1) readers done
    if (k0 + TKK < K) {   // overlap tile k+1 copy with the WMMAs on tile k
      const int kn = k0 + TKK;
      async_g2l_b128(&As[p ^ 1][s0], Ab + kn + gA0);
      async_g2l_b128(&As[p ^ 1][s1], Ab + kn + gA1);
      async_g2l_b128(&Bs[p ^ 1][s0], Bb + kn + gA0);
      async_g2l_b128(&Bs[p ^ 1][s1], Bb + kn + gA1);
    }
    const v16bf a0 = load_frag(&As[p][(wr +  0) * LDSS], lane);
    const v16bf a1 = load_frag(&As[p][(wr + 16) * LDSS], lane);
    const v16bf b0 = load_frag(&Bs[p][(wc +  0) * LDSS], lane);
    const v16bf b1 = load_frag(&Bs[p][(wc + 16) * LDSS], lane);
    const v16bf b2 = load_frag(&Bs[p][(wc + 32) * LDSS], lane);
    const v16bf b3 = load_frag(&Bs[p][(wc + 48) * LDSS], lane);
    WMMA_BF16(c00, a0, b0); WMMA_BF16(c01, a0, b1);
    WMMA_BF16(c02, a0, b2); WMMA_BF16(c03, a0, b3);
    WMMA_BF16(c10, a1, b0); WMMA_BF16(c11, a1, b1);
    WMMA_BF16(c12, a1, b2); WMMA_BF16(c13, a1, b3);
    p ^= 1;
  }
#else
  // ---- register-staged pipeline (proven fallback)
  uint4 fa0 = ld16g(Ab + gA0);
  uint4 fa1 = ld16g(Ab + gA1);
  uint4 fb0 = ld16g(Bb + gA0);
  uint4 fb1 = ld16g(Bb + gA1);

  int p = 0;
  for (int k0 = 0; k0 < K; k0 += TKK) {
    st16s(&As[p][s0], fa0);
    st16s(&As[p][s1], fa1);
    st16s(&Bs[p][s0], fb0);
    st16s(&Bs[p][s1], fb1);
    __syncthreads();
    if (k0 + TKK < K) {
      const int kn = k0 + TKK;
      fa0 = ld16g(Ab + kn + gA0);
      fa1 = ld16g(Ab + kn + gA1);
      fb0 = ld16g(Bb + kn + gA0);
      fb1 = ld16g(Bb + kn + gA1);
    }
    const v16bf a0 = load_frag(&As[p][(wr +  0) * LDSS], lane);
    const v16bf a1 = load_frag(&As[p][(wr + 16) * LDSS], lane);
    const v16bf b0 = load_frag(&Bs[p][(wc +  0) * LDSS], lane);
    const v16bf b1 = load_frag(&Bs[p][(wc + 16) * LDSS], lane);
    const v16bf b2 = load_frag(&Bs[p][(wc + 32) * LDSS], lane);
    const v16bf b3 = load_frag(&Bs[p][(wc + 48) * LDSS], lane);
    WMMA_BF16(c00, a0, b0); WMMA_BF16(c01, a0, b1);
    WMMA_BF16(c02, a0, b2); WMMA_BF16(c03, a0, b3);
    WMMA_BF16(c10, a1, b0); WMMA_BF16(c11, a1, b1);
    WMMA_BF16(c12, a1, b2); WMMA_BF16(c13, a1, b3);
    p ^= 1;
  }
#endif

  store_tile<EPI, OUTBF>(Cout, bias, c00, m0 + wr +  0, n0 + wc +  0, N, lane);
  store_tile<EPI, OUTBF>(Cout, bias, c01, m0 + wr +  0, n0 + wc + 16, N, lane);
  store_tile<EPI, OUTBF>(Cout, bias, c02, m0 + wr +  0, n0 + wc + 32, N, lane);
  store_tile<EPI, OUTBF>(Cout, bias, c03, m0 + wr +  0, n0 + wc + 48, N, lane);
  store_tile<EPI, OUTBF>(Cout, bias, c10, m0 + wr + 16, n0 + wc +  0, N, lane);
  store_tile<EPI, OUTBF>(Cout, bias, c11, m0 + wr + 16, n0 + wc + 16, N, lane);
  store_tile<EPI, OUTBF>(Cout, bias, c12, m0 + wr + 16, n0 + wc + 32, N, lane);
  store_tile<EPI, OUTBF>(Cout, bias, c13, m0 + wr + 16, n0 + wc + 48, N, lane);
}

// ---- elementwise fp32 -> bf16 conversion (8 elements / thread) -------------
__global__ __launch_bounds__(256) void cvt_f32_bf16_kernel(
    const float* __restrict__ in, __bf16* __restrict__ out, size_t n8) {
  const size_t i = (size_t)blockIdx.x * 256 + threadIdx.x;
  if (i >= n8) return;
  const float4 f0 = reinterpret_cast<const float4*>(in)[i * 2];
  const float4 f1 = reinterpret_cast<const float4*>(in)[i * 2 + 1];
  union { __bf16 h[8]; uint4 u; } pk;
  pk.h[0] = (__bf16)f0.x; pk.h[1] = (__bf16)f0.y;
  pk.h[2] = (__bf16)f0.z; pk.h[3] = (__bf16)f0.w;
  pk.h[4] = (__bf16)f1.x; pk.h[5] = (__bf16)f1.y;
  pk.h[6] = (__bf16)f1.z; pk.h[7] = (__bf16)f1.w;
  reinterpret_cast<uint4*>(out)[i] = pk.u;
}

// Causal depthwise conv (taps=4) + bias + SiLU. in: bf16 [B,L,D];
// writes fp32 (for the scan) and bf16 (for the delta GEMM).
__global__ __launch_bounds__(256) void conv_silu_kernel(
    const __bf16* __restrict__ in, const float* __restrict__ w,
    const float* __restrict__ b, float* __restrict__ outf,
    __bf16* __restrict__ outh, int Ln, int D, size_t total) {
  const size_t i = (size_t)blockIdx.x * 256 + threadIdx.x;
  if (i >= total) return;
  const int d = (int)(i % D);
  const int t = (int)((i / D) % Ln);
  const float* w4 = &w[d * 4];
  float acc = b[d];
#pragma unroll
  for (int j = 0; j < 4; ++j) {
    const int tt = t - 3 + j;
    if (tt >= 0) acc += w4[j] * (float)in[i + (size_t)(j - 3) * D];
  }
  const float v = silu_f(acc);
  outf[i] = v;
  outh[i] = (__bf16)v;
}

// Bm[row,n] = left[row,:]·W_B[n,:], Cm likewise.  One block per row (D=2048).
__global__ __launch_bounds__(256) void bc_kernel(
    const float* __restrict__ left, const float* __restrict__ WB,
    const float* __restrict__ WC, float* __restrict__ Bm,
    float* __restrict__ Cm, int D) {
  __shared__ float srow[2048];
  __shared__ float part[32][9];
  const int row = blockIdx.x;
  const int tid = threadIdx.x;
  for (int c = 0; c < D / 256; ++c)
    srow[tid + c * 256] = left[(size_t)row * D + tid + c * 256];
  __syncthreads();
  const int n     = tid & 31;   // output index (0..15 -> B, 16..31 -> C)
  const int slice = tid >> 5;   // k slice 0..7
  const float* wrow = (n < 16) ? &WB[n * D] : &WC[(n - 16) * D];
  const int kpart = D / 8, kk0 = slice * kpart;
  float acc = 0.0f;
  for (int k = 0; k < kpart; ++k) acc += srow[kk0 + k] * wrow[kk0 + k];
  part[n][slice] = acc;
  __syncthreads();
  if (tid < 32) {
    float s = 0.0f;
#pragma unroll
    for (int j = 0; j < 8; ++j) s += part[tid][j];
    if (tid < 16) Bm[(size_t)row * 16 + tid]        = s;
    else          Cm[(size_t)row * 16 + (tid - 16)] = s;
  }
}

// Selective scan: each lane owns one channel d of one batch b; h[16] in VGPRs.
// B_t / C_t broadcast via LDS, double-buffered (one barrier per step).
__global__ __launch_bounds__(256) void scan_kernel(
    const float* __restrict__ delta, const float* __restrict__ xin,
    const float* __restrict__ Bm, const float* __restrict__ Cm,
    const float* __restrict__ A_log, float* __restrict__ y,
    int Ln, int D) {
  const int blocksPerB = D / 256;
  const int b    = blockIdx.x / blocksPerB;
  const int dblk = blockIdx.x % blocksPerB;
  const int tid  = threadIdx.x;
  const int d    = dblk * 256 + tid;

  float A[16], h[16];
#pragma unroll
  for (int n = 0; n < 16; ++n) {
    A[n] = -__logf(1.0f + __expf(A_log[n]));
    h[n] = 0.0f;
  }

  __shared__ float sB[2][16], sC[2][16];
  const size_t row0 = (size_t)b * Ln;
  if (tid < 16)       sB[0][tid]      = Bm[row0 * 16 + tid];
  else if (tid < 32)  sC[0][tid - 16] = Cm[row0 * 16 + (tid - 16)];

  for (int t = 0; t < Ln; ++t) {
    __syncthreads();
    const int cur = t & 1;
    const size_t row = row0 + t;
    const float dt = delta[row * D + d];
    const float xt = xin[row * D + d];
    if (t + 1 < Ln) {
      if (tid < 16)      sB[cur ^ 1][tid]      = Bm[(row + 1) * 16 + tid];
      else if (tid < 32) sC[cur ^ 1][tid - 16] = Cm[(row + 1) * 16 + (tid - 16)];
    }
    const float dx = dt * xt;
    float acc = 0.0f;
#pragma unroll
    for (int n = 0; n < 16; ++n) {
      h[n] = __expf(dt * A[n]) * h[n] + dx * sB[cur][n];
      acc += h[n] * sC[cur][n];
    }
    y[row * D + d] = acc;
  }
}

// y_ln = LN(ssm * gate) * g + b, written as bf16 for the final WMMA GEMM.
__global__ __launch_bounds__(256) void gate_ln_kernel(
    const float* __restrict__ ssm, const __bf16* __restrict__ gate,
    const float* __restrict__ g, const float* __restrict__ bta,
    __bf16* __restrict__ out, int D) {
  __shared__ float rs[256], rss[256];
  const int row = blockIdx.x;
  const int tid = threadIdx.x;
  float v[8];
  float s = 0.0f, ss = 0.0f;
#pragma unroll
  for (int i = 0; i < 8; ++i) {
    const int dd = tid + i * 256;
    const float x = ssm[(size_t)row * D + dd] * (float)gate[(size_t)row * D + dd];
    v[i] = x; s += x; ss += x * x;
  }
  rs[tid] = s; rss[tid] = ss;
  __syncthreads();
  for (int o = 128; o > 0; o >>= 1) {
    if (tid < o) { rs[tid] += rs[tid + o]; rss[tid] += rss[tid + o]; }
    __syncthreads();
  }
  const float mu  = rs[0] / (float)D;
  const float var = rss[0] / (float)D - mu * mu;
  const float inv = rsqrtf(var + 1e-5f);
#pragma unroll
  for (int i = 0; i < 8; ++i) {
    const int dd = tid + i * 256;
    out[(size_t)row * D + dd] = (__bf16)((v[i] - mu) * inv * g[dd] + bta[dd]);
  }
}

// ---------------------------------------------------------------------------

extern "C" void kernel_launch(void* const* d_in, const int* in_sizes, int n_in,
                              void* d_out, int out_size, void* d_ws, size_t ws_size,
                              hipStream_t stream) {
  (void)in_sizes; (void)n_in; (void)out_size; (void)ws_size;
  const float* x       = (const float*)d_in[0];
  const float* W_left  = (const float*)d_in[1];
  const float* conv_w  = (const float*)d_in[2];
  const float* conv_b  = (const float*)d_in[3];
  const float* W_delta = (const float*)d_in[4];
  const float* b_delta = (const float*)d_in[5];
  const float* W_B     = (const float*)d_in[6];
  const float* W_C     = (const float*)d_in[7];
  const float* A_log   = (const float*)d_in[8];
  const float* W_right = (const float*)d_in[9];
  const float* ln_g    = (const float*)d_in[10];
  const float* ln_b    = (const float*)d_in[11];
  const float* W_out   = (const float*)d_in[12];
  float* out = (float*)d_out;

  const int Bn = 4, Ln = 2048, Dm = 1024, Di = 2048;
  const int Mrows = Bn * Ln;                  // 8192
  const size_t big = (size_t)Mrows * Di;      // 16.7M elements

  // ---- workspace layout ----
  char* w8 = (char*)d_ws;
  float*  left_f  = (float*)w8;            w8 += big * 4;             // conv out (scan x_t)
  float*  delta_f = (float*)w8;            w8 += big * 4;             // softplus-clipped delta
  float*  ssm_f   = (float*)w8;            w8 += big * 4;             // scan output
  float*  Bm      = (float*)w8;            w8 += (size_t)Mrows * 16 * 4;
  float*  Cm      = (float*)w8;            w8 += (size_t)Mrows * 16 * 4;
  __bf16* x_bf    = (__bf16*)w8;           w8 += (size_t)Mrows * Dm * 2;
  __bf16* Wl_bf   = (__bf16*)w8;           w8 += (size_t)Di * Dm * 2;
  __bf16* Wr_bf   = (__bf16*)w8;           w8 += (size_t)Di * Dm * 2;
  __bf16* Wd_bf   = (__bf16*)w8;           w8 += (size_t)Di * Di * 2;
  __bf16* Wo_bf   = (__bf16*)w8;           w8 += (size_t)Dm * Di * 2;
  __bf16* lp_bf   = (__bf16*)w8;           w8 += big * 2;             // left_pre; reused as y_ln
  __bf16* left_bf = (__bf16*)w8;           w8 += big * 2;             // conv out (delta GEMM A)
  __bf16* gate_bf = (__bf16*)w8;           w8 += big * 2;             // silu(x @ W_right^T)
  __bf16* yln_bf  = lp_bf;                                            // reuse (dead after conv)

  const dim3 blk(256);
  const dim3 gInner(Mrows / TM, Di / TN);   // 64 x 16
  const dim3 gOut(Mrows / TM, Dm / TN);     // 64 x 8

  // ---- 0) one-time fp32 -> bf16 conversions (x + the 4 GEMM weights) ----
  auto cvt = [&](const float* src, __bf16* dst, size_t n) {
    const size_t n8 = n / 8;
    cvt_f32_bf16_kernel<<<(unsigned)((n8 + 255) / 256), blk, 0, stream>>>(src, dst, n8);
  };
  cvt(x,       x_bf, (size_t)Mrows * Dm);
  cvt(W_left,  Wl_bf, (size_t)Di * Dm);
  cvt(W_right, Wr_bf, (size_t)Di * Dm);
  cvt(W_delta, Wd_bf, (size_t)Di * Di);
  cvt(W_out,   Wo_bf, (size_t)Dm * Di);

  // 1) left_pre = x @ W_left^T           (bf16 out, consumed by conv)
  gemm_bf16_wmma<0, 1><<<gInner, blk, 0, stream>>>(x_bf, Wl_bf, nullptr, lp_bf, Mrows, Di, Dm);
  // 2) gate = silu(x @ W_right^T)        (silu fused, bf16 out)
  gemm_bf16_wmma<2, 1><<<gInner, blk, 0, stream>>>(x_bf, Wr_bf, nullptr, gate_bf, Mrows, Di, Dm);
  // 3) left = silu(causal_dwconv(left_pre) + conv_b)  -> fp32 + bf16
  conv_silu_kernel<<<(unsigned)((big + 255) / 256), blk, 0, stream>>>(
      lp_bf, conv_w, conv_b, left_f, left_bf, Ln, Di, big);
  // 4) delta = clip(softplus(left @ W_delta^T + b_delta))  (fused, fp32 out)
  gemm_bf16_wmma<1, 0><<<gInner, blk, 0, stream>>>(left_bf, Wd_bf, b_delta, delta_f, Mrows, Di, Di);
  // 5) Bm, Cm projections (N = 16 each)
  bc_kernel<<<Mrows, blk, 0, stream>>>(left_f, W_B, W_C, Bm, Cm, Di);
  // 6) selective scan -> ssm_f
  scan_kernel<<<Bn * (Di / 256), blk, 0, stream>>>(delta_f, left_f, Bm, Cm, A_log, ssm_f, Ln, Di);
  // 7) gate multiply + layernorm -> y_ln (bf16, reuses left_pre buffer)
  gate_ln_kernel<<<Mrows, blk, 0, stream>>>(ssm_f, gate_bf, ln_g, ln_b, yln_bf, Di);
  // 8) out = y_ln @ W_out^T (fp32 out)
  gemm_bf16_wmma<0, 0><<<gOut, blk, 0, stream>>>(yln_bf, Wo_bf, nullptr, out, Mrows, Dm, Di);
}
